// RWKV5TimeMix_80599356277343
// MI455X (gfx1250) — compile-verified, matrix operand order checked by reading, not run on hardware
//
#include <hip/hip_runtime.h>

typedef __bf16 v16bf __attribute__((ext_vector_type(16)));
typedef float  v8f   __attribute__((ext_vector_type(8)));
typedef unsigned int v4u __attribute__((ext_vector_type(4)));
typedef unsigned short u16;

static constexpr int Bb = 4;
static constexpr int Tt = 4096;
static constexpr int Cc = 2048;
static constexpr int Hh = 32;
static constexpr int Kk = 64;
static constexpr int Ll = 64;      // chunk length
static constexpr int Mrows = Bb * Tt;   // 16384

// ---------- bf16 helpers (round-to-nearest-even) ----------
__device__ __forceinline__ u16 f2bf(float x) {
  unsigned u = __float_as_uint(x);
  unsigned r = u + 0x7FFFu + ((u >> 16) & 1u);
  return (u16)(r >> 16);
}
__device__ __forceinline__ float bf2f(u16 s) {
  return __uint_as_float(((unsigned)s) << 16);
}

// ---------- async global->LDS (CDNA5 GLOBAL_LOAD_ASYNC_TO_LDS_B128) ----------
// Per ISA 10.x: each lane copies 16B from global memory into LDS (offset added on both sides),
// tracked by ASYNCcnt. Generic LDS pointer low 32 bits == LDS offset.
__device__ __forceinline__ void async_copy_64B(void* lds_dst, const void* gsrc) {
  unsigned ldsoff = (unsigned)(size_t)lds_dst;
  unsigned long long ga = (unsigned long long)(size_t)gsrc;
  asm volatile(
      "global_load_async_to_lds_b128 %0, %1, off\n\t"
      "global_load_async_to_lds_b128 %0, %1, off offset:16\n\t"
      "global_load_async_to_lds_b128 %0, %1, off offset:32\n\t"
      "global_load_async_to_lds_b128 %0, %1, off offset:48"
      :: "v"(ldsoff), "v"(ga) : "memory");
}
__device__ __forceinline__ void async_wait0() {
  asm volatile("s_wait_asynccnt 0x0" ::: "memory");
}

// ---------- WMMA fragment loader ----------
// ISA 16-bit A/Bt layout (wave32): lane l<16 : row=l, halves0-7 = K[k0..k0+7],
// halves8-15 = K[k0+16..k0+23]; lane l>=16: row=l-16, K groups {8..15},{24..31}.
__device__ __forceinline__ v16bf ld_frag(const u16* p, int row0, int k0, int lane, int stride) {
  const u16* r = p + (size_t)(row0 + (lane & 15)) * stride + k0 + ((lane >> 4) << 3);
  union { v16bf v; v4u q[2]; } u;
  u.q[0] = *(const v4u*)(r);
  u.q[1] = *(const v4u*)(r + 16);
  return u.v;
}

__device__ __forceinline__ v8f wmma_bf16(v16bf a, v16bf b, v8f c) {
  return __builtin_amdgcn_wmma_f32_16x16x32_bf16(false, a, false, b, (short)0, c, false, false);
}

// =====================================================================
// Kernel 1: convert + TRANSPOSE 5 weight matrices f32 -> bf16 W^T [n][k]
// (so GEMM B tiles are contiguous along k and can use async b128 -> LDS)
// =====================================================================
__global__ __launch_bounds__(256) void wtrans_kernel(
    const float* __restrict__ wr, const float* __restrict__ wk,
    const float* __restrict__ wv, const float* __restrict__ wg,
    const float* __restrict__ wo, u16* __restrict__ dst) {
  const int q = blockIdx.z;
  const float* src = (q == 0) ? wr : (q == 1) ? wk : (q == 2) ? wv : (q == 3) ? wg : wo;
  u16* d = dst + (size_t)q * Cc * Cc;

  __shared__ u16 Tl[64][65];
  const int tid = threadIdx.x;
  const int k0 = blockIdx.x * 64, n0 = blockIdx.y * 64;
  {
    int kk = tid >> 2, nn0 = (tid & 3) * 16;
    const float* s = src + (size_t)(k0 + kk) * Cc + n0 + nn0;
#pragma unroll
    for (int i = 0; i < 16; ++i) Tl[kk][nn0 + i] = f2bf(s[i]);
  }
  __syncthreads();
  {
    int nn = tid >> 2, kk0 = (tid & 3) * 16;
    u16* dd = d + (size_t)(n0 + nn) * Cc + k0 + kk0;
#pragma unroll
    for (int i = 0; i < 16; ++i) dd[i] = Tl[kk0 + i][nn];
  }
}

// =====================================================================
// Kernel 2: batched mixed-input GEMM (z = 0..3 -> r,k,v,g_pre), bf16 WMMA
//   A[m,k] = x[m,k]*mix[k] + xprev[m,k]*(1-mix[k])   (token shift on the fly)
//   out = A @ W  (bf16 out); B tile via async global->LDS from W^T
// =====================================================================
static constexpr int BM = 128, BN = 256, BK = 32, LDA = BK + 8;

__global__ __launch_bounds__(256) void gemm_mix_kernel(
    const float* __restrict__ x, const float* __restrict__ shift,
    const float* __restrict__ mr, const float* __restrict__ mk,
    const float* __restrict__ mv, const float* __restrict__ mg,
    const u16* __restrict__ WTbf,
    u16* __restrict__ rb, u16* __restrict__ kbuf,
    u16* __restrict__ vbuf, u16* __restrict__ gbuf) {
  const int z = blockIdx.z;
  const float* mix = (z == 0) ? mr : (z == 1) ? mk : (z == 2) ? mv : mg;
  const u16* WT = WTbf + (size_t)z * Cc * Cc;          // W^T: [n][k]
  u16* dst = (z == 0) ? rb : (z == 1) ? kbuf : (z == 2) ? vbuf : gbuf;

  __shared__ __align__(16) u16 As[BM * LDA];
  __shared__ __align__(16) u16 Bs[BN * LDA];

  const int tid = threadIdx.x;
  const int lane = tid & 31;
  const int wid = tid >> 5;
  const int wm = (wid & 1) * 64;          // 2x4 wave grid, 64x64 per wave
  const int wn = (wid >> 1) * 64;
  const int mBase = blockIdx.x * BM;
  const int nBase = blockIdx.y * BN;

  v8f acc[4][4];
#pragma unroll
  for (int i = 0; i < 4; ++i)
#pragma unroll
    for (int j = 0; j < 4; ++j) acc[i][j] = {};

  // A-fill indices: thread -> (row, 16-col group)
  const int ar = tid >> 1;
  const int ac = (tid & 1) * 16;
  const int gm = mBase + ar;
  const int tpos = gm & (Tt - 1);
  const int bidx = gm >> 12;

  for (int k0 = 0; k0 < Cc; k0 += BK) {
    __syncthreads();
    // B tile: row n = nBase+tid, 32 contiguous k halves (64B) -> async to LDS
    {
      const u16* wsrc = WT + (size_t)(nBase + tid) * Cc + k0;
      __builtin_prefetch(wsrc + BK, 0, 1);   // next k-panel (contiguous in W^T)
      async_copy_64B(&Bs[tid * LDA], wsrc);
    }
    // A tile: token-shift mix + f32->bf16, two b128 LDS stores
    {
      const float* cur = x + (size_t)gm * Cc + k0 + ac;
      const float* prv = (tpos == 0) ? (shift + (size_t)bidx * Cc + k0 + ac)
                                     : (x + (size_t)(gm - 1) * Cc + k0 + ac);
      union { v4u q[2]; u16 s[16]; } uu;
#pragma unroll
      for (int i = 0; i < 16; ++i) {
        float m = mix[k0 + ac + i];
        uu.s[i] = f2bf(cur[i] * m + prv[i] * (1.0f - m));
      }
      *(v4u*)(&As[ar * LDA + ac])     = uu.q[0];
      *(v4u*)(&As[ar * LDA + ac + 8]) = uu.q[1];
    }
    async_wait0();
    __syncthreads();

    v16bf afr[4];
#pragma unroll
    for (int tm = 0; tm < 4; ++tm) afr[tm] = ld_frag(As, wm + tm * 16, 0, lane, LDA);
#pragma unroll
    for (int tn = 0; tn < 4; ++tn) {
      v16bf bfr = ld_frag(Bs, wn + tn * 16, 0, lane, LDA);
#pragma unroll
      for (int tm = 0; tm < 4; ++tm) acc[tm][tn] = wmma_bf16(afr[tm], bfr, acc[tm][tn]);
    }
  }

  const int rlo = ((lane >> 4) << 3);
  const int cl = lane & 15;
#pragma unroll
  for (int tm = 0; tm < 4; ++tm)
#pragma unroll
    for (int tn = 0; tn < 4; ++tn)
#pragma unroll
      for (int i = 0; i < 8; ++i) {
        int row = mBase + wm + tm * 16 + i + rlo;
        int col = nBase + wn + tn * 16 + cl;
        dst[(size_t)row * Cc + col] = f2bf(acc[tm][tn][i]);
      }
}

// =====================================================================
// Kernel 3: WKV chunked scan, one workgroup per (b,h), bf16 WMMA inner mm
// =====================================================================
__global__ __launch_bounds__(256) void wkv_kernel(
    const u16* __restrict__ rbuf, const u16* __restrict__ kbuf,
    const u16* __restrict__ vbuf, const float* __restrict__ td,
    const float* __restrict__ tf, const float* __restrict__ S0,
    u16* __restrict__ ybuf, float* __restrict__ Sout) {
  const int bh = blockIdx.x;
  const int b = bh >> 5;
  const int h = bh & 31;
  const int tid = threadIdx.x;
  const int lane = tid & 31;
  const int wid = tid >> 5;

  __shared__ float S[Kk * Kk];        // state, f32
  __shared__ float diag[Ll];
  __shared__ float lw[Kk], uu[Kk];
  __shared__ __align__(16) u16 rwb[Ll * Kk];    // rw[t][k]
  __shared__ __align__(16) u16 kwb[Ll * Kk];    // kw[s][k], reused as A[t][s]
  __shared__ __align__(16) u16 kremT[Kk * Ll];  // krem^T [k][s]
  __shared__ __align__(16) u16 vcT[Kk * Ll];    // v^T [j][s]
  __shared__ __align__(16) u16 SbT[Kk * Kk];    // bf16 S^T snapshot [j][k]

  if (tid < Kk) {
    lw[tid] = -__expf(td[h * Kk + tid]) * 1.44269504088896340736f;  // log2(w)
    uu[tid] = tf[h * Kk + tid];
  }
  for (int e = tid; e < Kk * Kk; e += 256) S[e] = S0[(size_t)bh * Kk * Kk + e];
  __syncthreads();

  const int srow = tid >> 2;
  const int cg = (tid & 3) * 16;
  const int rlo = ((lane >> 4) << 3);
  const int cl = lane & 15;

  for (int ch = 0; ch < Tt / Ll; ++ch) {
    // snapshot S -> bf16 transposed; zero diag
    if (tid < Ll) diag[tid] = 0.0f;
    for (int e = tid; e < Kk * Kk; e += 256) {
      int j = e >> 6, kk2 = e & 63;
      SbT[e] = f2bf(S[kk2 * Kk + j]);
    }
    __syncthreads();

    // phase 1: decay-scaled operands
    {
      float sf = (float)srow;
      size_t gbase = ((size_t)(b * Tt + ch * Ll + srow) * Hh + h) * Kk + cg;
      float dp = 0.0f;
#pragma unroll
      for (int i = 0; i < 16; ++i) {
        int kc = cg + i;
        float rc = bf2f(rbuf[gbase + i]);
        float kv = bf2f(kbuf[gbase + i]);
        float vv = bf2f(vbuf[gbase + i]);
        float l = lw[kc];
        float wt = exp2f(l * sf);
        float wi = exp2f(-l * (sf + 1.0f));
        float wr = exp2f(l * (63.0f - sf));
        rwb[srow * Kk + kc] = f2bf(rc * wt);
        kwb[srow * Kk + kc] = f2bf(kv * wi);
        kremT[kc * Ll + srow] = f2bf(kv * wr);
        vcT[kc * Ll + srow] = f2bf(vv);
        dp += rc * uu[kc] * kv;
      }
      atomicAdd(&diag[srow], dp);
    }
    __syncthreads();

    // phase 2: A = (rw @ kw^T) * strict-lower mask   (2 tiles per wave)
    v8f pa[2];
#pragma unroll
    for (int tt = 0; tt < 2; ++tt) {
      int tile = wid * 2 + tt;
      int tm = (tile >> 2) * 16, tn = (tile & 3) * 16;
      v8f acc = {};
#pragma unroll
      for (int ks = 0; ks < 2; ++ks)
        acc = wmma_bf16(ld_frag(rwb, tm, ks * 32, lane, Kk),
                        ld_frag(kwb, tn, ks * 32, lane, Kk), acc);
      pa[tt] = acc;
    }
    __syncthreads();
#pragma unroll
    for (int tt = 0; tt < 2; ++tt) {
      int tile = wid * 2 + tt;
      int tm = (tile >> 2) * 16, tn = (tile & 3) * 16;
#pragma unroll
      for (int i = 0; i < 8; ++i) {
        int tr = tm + i + rlo, sc = tn + cl;
        kwb[tr * Kk + sc] = f2bf((sc < tr) ? pa[tt][i] : 0.0f);  // kwb is now A
      }
    }
    __syncthreads();

    // phase 3: y = A@v + rw@S + diag*v
#pragma unroll
    for (int tt = 0; tt < 2; ++tt) {
      int tile = wid * 2 + tt;
      int tm = (tile >> 2) * 16, tn = (tile & 3) * 16;
      v8f acc = {};
#pragma unroll
      for (int ks = 0; ks < 2; ++ks)
        acc = wmma_bf16(ld_frag(kwb, tm, ks * 32, lane, Kk),
                        ld_frag(vcT, tn, ks * 32, lane, Kk), acc);
#pragma unroll
      for (int ks = 0; ks < 2; ++ks)
        acc = wmma_bf16(ld_frag(rwb, tm, ks * 32, lane, Kk),
                        ld_frag(SbT, tn, ks * 32, lane, Kk), acc);
#pragma unroll
      for (int i = 0; i < 8; ++i) {
        int tr = tm + i + rlo, jc = tn + cl;
        float y = acc[i] + diag[tr] * bf2f(vcT[jc * Ll + tr]);
        ybuf[((size_t)(b * Tt + ch * Ll + tr) * Hh + h) * Kk + jc] = f2bf(y);
      }
    }

    // phase 4: S = w^L * S + krem^T @ v
#pragma unroll
    for (int tt = 0; tt < 2; ++tt) {
      int tile = wid * 2 + tt;
      int km = (tile >> 2) * 16, jn = (tile & 3) * 16;
      v8f acc;
#pragma unroll
      for (int i = 0; i < 8; ++i) {
        int kr2 = km + i + rlo, jc = jn + cl;
        acc[i] = exp2f(lw[kr2] * 64.0f) * S[kr2 * Kk + jc];
      }
#pragma unroll
      for (int ks = 0; ks < 2; ++ks)
        acc = wmma_bf16(ld_frag(kremT, km, ks * 32, lane, Ll),
                        ld_frag(vcT, jn, ks * 32, lane, Ll), acc);
#pragma unroll
      for (int i = 0; i < 8; ++i) {
        int kr2 = km + i + rlo, jc = jn + cl;
        S[kr2 * Kk + jc] = acc[i];
      }
    }
    __syncthreads();
  }

  for (int e = tid; e < Kk * Kk; e += 256) Sout[(size_t)bh * Kk * Kk + e] = S[e];
}

// =====================================================================
// Kernel 4: per-(b,t) head group-norm of y/8 + SiLU(g) gating -> bf16
// =====================================================================
__global__ __launch_bounds__(256) void gnorm_gate_kernel(
    const u16* __restrict__ ybuf, const u16* __restrict__ gbuf,
    const float* __restrict__ lnw, const float* __restrict__ lnb,
    u16* __restrict__ ygbuf) {
  const size_t base = (size_t)blockIdx.x * Cc;
  const int tid = threadIdx.x;
  const int hh = tid >> 3;        // head
  const int k0 = (tid & 7) * 8;   // 8 elems per thread

  float v[8], sum = 0.0f, sq = 0.0f;
#pragma unroll
  for (int i = 0; i < 8; ++i) {
    v[i] = bf2f(ybuf[base + hh * Kk + k0 + i]) * 0.125f;  // / HEAD_DIV
    sum += v[i];
    sq += v[i] * v[i];
  }
  __shared__ float s1[256], s2[256];
  __shared__ float mh[Hh], rh[Hh];
  s1[tid] = sum; s2[tid] = sq;
  __syncthreads();
  if ((tid & 7) == 0) {
    float ts = 0.0f, tq = 0.0f;
#pragma unroll
    for (int j = 0; j < 8; ++j) { ts += s1[hh * 8 + j]; tq += s2[hh * 8 + j]; }
    float mu = ts * (1.0f / 64.0f);
    float var = tq * (1.0f / 64.0f) - mu * mu;
    mh[hh] = mu;
    rh[hh] = rsqrtf(var + 1e-5f);
  }
  __syncthreads();
#pragma unroll
  for (int i = 0; i < 8; ++i) {
    int kk2 = hh * Kk + k0 + i;
    float yn = (v[i] - mh[hh]) * rh[hh] * lnw[kk2] + lnb[kk2];
    float g = bf2f(gbuf[base + kk2]);
    g = g / (1.0f + __expf(-g));  // silu
    ygbuf[base + kk2] = f2bf(yn * g);
  }
}

// =====================================================================
// Kernel 5: final GEMM  out = yg @ Wo   (bf16 in, f32 out), async B tile
// =====================================================================
__global__ __launch_bounds__(256) void gemm_out_kernel(
    const u16* __restrict__ A, const u16* __restrict__ WT, float* __restrict__ out) {
  __shared__ __align__(16) u16 As[BM * LDA];
  __shared__ __align__(16) u16 Bs[BN * LDA];

  const int tid = threadIdx.x;
  const int lane = tid & 31;
  const int wid = tid >> 5;
  const int wm = (wid & 1) * 64;
  const int wn = (wid >> 1) * 64;
  const int mBase = blockIdx.x * BM;
  const int nBase = blockIdx.y * BN;

  v8f acc[4][4];
#pragma unroll
  for (int i = 0; i < 4; ++i)
#pragma unroll
    for (int j = 0; j < 4; ++j) acc[i][j] = {};

  const int ar = tid >> 1, ac = (tid & 1) * 16;
  const int gm = mBase + ar;

  for (int k0 = 0; k0 < Cc; k0 += BK) {
    __syncthreads();
    {
      const u16* wsrc = WT + (size_t)(nBase + tid) * Cc + k0;
      __builtin_prefetch(wsrc + BK, 0, 1);
      async_copy_64B(&Bs[tid * LDA], wsrc);
    }
    {
      const u16* src = A + (size_t)gm * Cc + k0 + ac;
      *(v4u*)(&As[ar * LDA + ac])     = *(const v4u*)(src);
      *(v4u*)(&As[ar * LDA + ac + 8]) = *(const v4u*)(src + 8);
    }
    async_wait0();
    __syncthreads();

    v16bf afr[4];
#pragma unroll
    for (int tm = 0; tm < 4; ++tm) afr[tm] = ld_frag(As, wm + tm * 16, 0, lane, LDA);
#pragma unroll
    for (int tn = 0; tn < 4; ++tn) {
      v16bf bfr = ld_frag(Bs, wn + tn * 16, 0, lane, LDA);
#pragma unroll
      for (int tm = 0; tm < 4; ++tm) acc[tm][tn] = wmma_bf16(afr[tm], bfr, acc[tm][tn]);
    }
  }

  const int rlo = ((lane >> 4) << 3);
  const int cl = lane & 15;
#pragma unroll
  for (int tm = 0; tm < 4; ++tm)
#pragma unroll
    for (int tn = 0; tn < 4; ++tn)
#pragma unroll
      for (int i = 0; i < 8; ++i) {
        int row = mBase + wm + tm * 16 + i + rlo;
        int col = nBase + wn + tn * 16 + cl;
        out[(size_t)row * Cc + col] = acc[tm][tn][i];
      }
}

// =====================================================================
// Kernel 6: shift_state_out = x[:, -1]
// =====================================================================
__global__ __launch_bounds__(256) void shift_out_kernel(const float* __restrict__ x,
                                                        float* __restrict__ dst) {
  int idx = blockIdx.x * blockDim.x + threadIdx.x;
  if (idx >= Bb * Cc) return;
  int b = idx >> 11, c = idx & (Cc - 1);
  dst[idx] = x[((size_t)b * Tt + (Tt - 1)) * Cc + c];
}

// =====================================================================
extern "C" void kernel_launch(void* const* d_in, const int* in_sizes, int n_in,
                              void* d_out, int out_size, void* d_ws, size_t ws_size,
                              hipStream_t stream) {
  const float* x      = (const float*)d_in[0];
  const float* shin   = (const float*)d_in[1];
  const float* wkv0   = (const float*)d_in[2];
  const float* mix_k  = (const float*)d_in[3];
  const float* mix_v  = (const float*)d_in[4];
  const float* mix_r  = (const float*)d_in[5];
  const float* mix_g  = (const float*)d_in[6];
  const float* tdcy   = (const float*)d_in[7];
  const float* tfaa   = (const float*)d_in[8];
  const float* Wr     = (const float*)d_in[9];
  const float* Wk     = (const float*)d_in[10];
  const float* Wv     = (const float*)d_in[11];
  const float* Wg     = (const float*)d_in[12];
  const float* Wo     = (const float*)d_in[13];
  const float* ln_w   = (const float*)d_in[14];
  const float* ln_b   = (const float*)d_in[15];
  (void)in_sizes; (void)n_in; (void)out_size; (void)ws_size;

  float* out = (float*)d_out;
  float* shift_out = out + (size_t)Mrows * Cc;
  float* wkv_out = shift_out + (size_t)Bb * Cc;

  const size_t CC = (size_t)Cc * Cc;
  const size_t BTC = (size_t)Mrows * Cc;
  u16* WTbf = (u16*)d_ws;        // 5*CC (transposed bf16 weights)
  u16* rb   = WTbf + 5 * CC;     // BTC each
  u16* kb   = rb + BTC;
  u16* vb   = kb + BTC;
  u16* gb   = vb + BTC;
  u16* yb   = gb + BTC;
  u16* ygb  = yb + BTC;

  // 1) weights -> bf16, transposed
  wtrans_kernel<<<dim3(Cc / 64, Cc / 64, 5), 256, 0, stream>>>(Wr, Wk, Wv, Wg, Wo, WTbf);

  // 2) r,k,v,g_pre GEMMs with fused token-shift mixing
  gemm_mix_kernel<<<dim3(Mrows / BM, Cc / BN, 4), 256, 0, stream>>>(
      x, shin, mix_r, mix_k, mix_v, mix_g, WTbf, rb, kb, vb, gb);

  // 3) WKV chunked scan
  wkv_kernel<<<Bb * Hh, 256, 0, stream>>>(rb, kb, vb, tdcy, tfaa, wkv0, yb, wkv_out);

  // 4) group-norm + gate
  gnorm_gate_kernel<<<Mrows, 256, 0, stream>>>(yb, gb, ln_w, ln_b, ygb);

  // 5) output projection
  gemm_out_kernel<<<dim3(Mrows / BM, Cc / BN, 1), 256, 0, stream>>>(
      ygb, WTbf + 4 * CC, out);

  // 6) shift state out
  shift_out_kernel<<<(Bb * Cc + 255) / 256, 256, 0, stream>>>(x, shift_out);
}
